// Block_31138512896634
// MI455X (gfx1250) — matrix operand order repaired and checked by reading, not compiled, not run on hardware
//
#include <hip/hip_runtime.h>
#include <hip/hip_bf16.h>
#include <stdint.h>

#define T_SEQ 2048
#define C_DIM 768
#define H_HEADS 12
#define D_HEAD 64
#define C3 (3 * C_DIM)
#define C4 (4 * C_DIM)

typedef __attribute__((ext_vector_type(16))) _Float16 v16h;
typedef __attribute__((ext_vector_type(8)))  float    v8f;
typedef __attribute__((ext_vector_type(4)))  int      v4i;

union Frag {
    v16h v;
    _Float16 h[16];
    uint4 q[2];
};

__device__ __forceinline__ v8f wmma16(const v16h a, const v16h b, const v8f c) {
    return __builtin_amdgcn_wmma_f32_16x16x32_f16(false, a, false, b, (short)0, c,
                                                  false, false);
}

// ---- CDNA5 async global->LDS staging (guarded; falls back to ld+ds_store) ----
#if defined(__has_builtin)
#if __has_builtin(__builtin_amdgcn_global_load_async_to_lds_b128)
#define ASYNC_LDS 1
#endif
#endif

typedef v4i __attribute__((address_space(1)))* gas_v4i;
typedef v4i __attribute__((address_space(3)))* las_v4i;

__device__ __forceinline__ void stage16(const _Float16* __restrict__ g,
                                        _Float16* __restrict__ l) {
#ifdef ASYNC_LDS
    __builtin_amdgcn_global_load_async_to_lds_b128((gas_v4i)g, (las_v4i)l, 0, 0);
#else
    *(uint4*)l = *(const uint4*)g;
#endif
}

__device__ __forceinline__ void stage_wait() {
#ifdef ASYNC_LDS
#if __has_builtin(__builtin_amdgcn_s_wait_asynccnt)
    __builtin_amdgcn_s_wait_asynccnt(0);
#else
    asm volatile("s_wait_asynccnt 0x0" ::: "memory");
#endif
#endif
}

// ---------------------------------------------------------------- converts
__global__ void convert_f32_to_f16(const float* __restrict__ in,
                                   _Float16* __restrict__ out, int n) {
    for (int i = blockIdx.x * blockDim.x + threadIdx.x; i < n;
         i += gridDim.x * blockDim.x)
        out[i] = (_Float16)in[i];
}

// in: rows x cols (row-major)  ->  out: cols x rows (row-major, transposed)
__global__ void transpose_f32_to_f16(const float* __restrict__ in,
                                     _Float16* __restrict__ out, int rows, int cols) {
    int n = rows * cols;
    for (int i = blockIdx.x * blockDim.x + threadIdx.x; i < n;
         i += gridDim.x * blockDim.x) {
        int c = i / rows;      // out row
        int r = i - c * rows;  // out col
        out[i] = (_Float16)in[(size_t)r * cols + c];
    }
}

// ---------------------------------------------------------------- layernorm
__global__ __launch_bounds__(256) void ln_f16(const float* __restrict__ x,
                                              const float* __restrict__ w,
                                              const float* __restrict__ b,
                                              _Float16* __restrict__ out) {
    int t = blockIdx.x;
    const float* xr = x + (size_t)t * C_DIM;
    float s = 0.f, s2 = 0.f;
    for (int c = threadIdx.x; c < C_DIM; c += blockDim.x) {
        float v = xr[c];
        s += v; s2 += v * v;
    }
    for (int m = 16; m >= 1; m >>= 1) {
        s  += __shfl_xor(s, m, 32);
        s2 += __shfl_xor(s2, m, 32);
    }
    __shared__ float ss[8], ss2[8];
    int wave = threadIdx.x >> 5, lane = threadIdx.x & 31;
    if (lane == 0) { ss[wave] = s; ss2[wave] = s2; }
    __syncthreads();
    s = 0.f; s2 = 0.f;
    for (int i = 0; i < 8; i++) { s += ss[i]; s2 += ss2[i]; }
    float mean = s * (1.0f / C_DIM);
    float var = s2 * (1.0f / C_DIM) - mean * mean;
    float rstd = rsqrtf(var + 1e-5f);
    _Float16* outr = out + (size_t)t * C_DIM;
    for (int c = threadIdx.x; c < C_DIM; c += blockDim.x)
        outr[c] = (_Float16)((xr[c] - mean) * rstd * w[c] + b[c]);
}

// ---------------------------------------------------------------- WMMA GEMM
// out[m,n] = sum_k A[m,k]*B[n,k] + bias[n]  (+ resid[m,n] on f32 path)
// Block tile 128x128, 8 waves; each wave 32x64 -> 8 WMMAs per 32-K chunk.
__global__ __launch_bounds__(256) void gemm_wmma_f16(
    const _Float16* __restrict__ A, const _Float16* __restrict__ B,
    const float* __restrict__ bias, const float* __restrict__ resid,
    float* __restrict__ outf, _Float16* __restrict__ outh,
    int M, int N, int K) {
    __shared__ __align__(16) _Float16 As[128 * 32];
    __shared__ __align__(16) _Float16 Bs[128 * 32];
    const int lane = threadIdx.x & 31;
    const int wave = threadIdx.x >> 5;
    const int mBase = blockIdx.y * 128;
    const int nBase = blockIdx.x * 128;
    const int waveM = wave >> 1;   // 0..3 -> 32 rows each
    const int waveN = wave & 1;    // 0..1 -> 64 cols each
    const int r  = lane & 15;
    const int kh = lane >> 4;
    const int sRow = threadIdx.x >> 2;       // 0..63
    const int sK   = (threadIdx.x & 3) * 8;  // 0,8,16,24

    v8f acc[2][4];
    v8f zero = {};
    for (int i = 0; i < 2; i++)
        for (int j = 0; j < 4; j++) acc[i][j] = zero;

    for (int k0 = 0; k0 < K; k0 += 32) {
        __syncthreads();
        stage16(&A[(size_t)(mBase + sRow) * K + k0 + sK],      &As[sRow * 32 + sK]);
        stage16(&A[(size_t)(mBase + sRow + 64) * K + k0 + sK], &As[(sRow + 64) * 32 + sK]);
        stage16(&B[(size_t)(nBase + sRow) * K + k0 + sK],      &Bs[sRow * 32 + sK]);
        stage16(&B[(size_t)(nBase + sRow + 64) * K + k0 + sK], &Bs[(sRow + 64) * 32 + sK]);
        if (k0 + 32 < K) {
            __builtin_prefetch(&A[(size_t)(mBase + sRow) * K + k0 + 32 + sK], 0, 1);
            __builtin_prefetch(&B[(size_t)(nBase + sRow) * K + k0 + 32 + sK], 0, 1);
        }
        stage_wait();
        __syncthreads();

        Frag a0, a1;
        const int ar = 32 * waveM + r;
        a0.q[0] = *(const uint4*)&As[ar * 32 + kh * 8];
        a0.q[1] = *(const uint4*)&As[ar * 32 + 16 + kh * 8];
        a1.q[0] = *(const uint4*)&As[(ar + 16) * 32 + kh * 8];
        a1.q[1] = *(const uint4*)&As[(ar + 16) * 32 + 16 + kh * 8];

        Frag b[4];
        for (int j = 0; j < 4; j++) {
            const int bc = 64 * waveN + 16 * j + r;
            b[j].q[0] = *(const uint4*)&Bs[bc * 32 + kh * 8];
            b[j].q[1] = *(const uint4*)&Bs[bc * 32 + 16 + kh * 8];
        }
        for (int j = 0; j < 4; j++) acc[0][j] = wmma16(a0.v, b[j].v, acc[0][j]);
        for (int j = 0; j < 4; j++) acc[1][j] = wmma16(a1.v, b[j].v, acc[1][j]);
    }

    const int rbase = mBase + 32 * waveM + 8 * kh;
    const int cbase = nBase + 64 * waveN + r;
    if (outh) {
        for (int i = 0; i < 2; i++)
            for (int j = 0; j < 4; j++) {
                int col = cbase + 16 * j;
                float bv = bias[col];
                for (int v = 0; v < 8; v++) {
                    int row = rbase + 16 * i + v;
                    outh[(size_t)row * N + col] = (_Float16)(acc[i][j][v] + bv);
                }
            }
    } else {
        for (int i = 0; i < 2; i++)
            for (int j = 0; j < 4; j++) {
                int col = cbase + 16 * j;
                float bv = bias[col];
                for (int v = 0; v < 8; v++) {
                    int row = rbase + 16 * i + v;
                    size_t idx = (size_t)row * N + col;
                    outf[idx] = acc[i][j][v] + bv + resid[idx];
                }
            }
    }
}

// ---------------------------------------------------------------- SwiGLU dual GEMM
// g[m,n] = silu(sum_k A*Bw) * (sum_k A*Bv)
// Block tile 128x64, 8 waves; each wave 32x32 dual -> 8 WMMAs per chunk.
__global__ __launch_bounds__(256) void gemm_swiglu_wmma(
    const _Float16* __restrict__ A, const _Float16* __restrict__ Bw,
    const _Float16* __restrict__ Bv, _Float16* __restrict__ out,
    int M, int N, int K) {
    __shared__ __align__(16) _Float16 As[128 * 32];
    __shared__ __align__(16) _Float16 Ws[64 * 32];
    __shared__ __align__(16) _Float16 Vs[64 * 32];
    const int lane = threadIdx.x & 31;
    const int wave = threadIdx.x >> 5;
    const int mBase = blockIdx.y * 128;
    const int nBase = blockIdx.x * 64;
    const int waveM = wave >> 1;   // 0..3 -> 32 rows
    const int waveN = wave & 1;    // 0..1 -> 32 cols
    const int r  = lane & 15;
    const int kh = lane >> 4;
    const int sRow = threadIdx.x >> 2;
    const int sK   = (threadIdx.x & 3) * 8;

    v8f accS[2][2], accT[2][2];
    v8f zero = {};
    for (int i = 0; i < 2; i++)
        for (int j = 0; j < 2; j++) { accS[i][j] = zero; accT[i][j] = zero; }

    for (int k0 = 0; k0 < K; k0 += 32) {
        __syncthreads();
        stage16(&A[(size_t)(mBase + sRow) * K + k0 + sK],      &As[sRow * 32 + sK]);
        stage16(&A[(size_t)(mBase + sRow + 64) * K + k0 + sK], &As[(sRow + 64) * 32 + sK]);
        stage16(&Bw[(size_t)(nBase + sRow) * K + k0 + sK],     &Ws[sRow * 32 + sK]);
        stage16(&Bv[(size_t)(nBase + sRow) * K + k0 + sK],     &Vs[sRow * 32 + sK]);
        if (k0 + 32 < K) {
            __builtin_prefetch(&Bw[(size_t)(nBase + sRow) * K + k0 + 32 + sK], 0, 1);
            __builtin_prefetch(&Bv[(size_t)(nBase + sRow) * K + k0 + 32 + sK], 0, 1);
        }
        stage_wait();
        __syncthreads();

        Frag a0, a1;
        const int ar = 32 * waveM + r;
        a0.q[0] = *(const uint4*)&As[ar * 32 + kh * 8];
        a0.q[1] = *(const uint4*)&As[ar * 32 + 16 + kh * 8];
        a1.q[0] = *(const uint4*)&As[(ar + 16) * 32 + kh * 8];
        a1.q[1] = *(const uint4*)&As[(ar + 16) * 32 + 16 + kh * 8];

        Frag w[2], vv[2];
        for (int j = 0; j < 2; j++) {
            const int bc = 32 * waveN + 16 * j + r;
            w[j].q[0]  = *(const uint4*)&Ws[bc * 32 + kh * 8];
            w[j].q[1]  = *(const uint4*)&Ws[bc * 32 + 16 + kh * 8];
            vv[j].q[0] = *(const uint4*)&Vs[bc * 32 + kh * 8];
            vv[j].q[1] = *(const uint4*)&Vs[bc * 32 + 16 + kh * 8];
        }
        for (int j = 0; j < 2; j++) {
            accS[0][j] = wmma16(a0.v, w[j].v,  accS[0][j]);
            accS[1][j] = wmma16(a1.v, w[j].v,  accS[1][j]);
            accT[0][j] = wmma16(a0.v, vv[j].v, accT[0][j]);
            accT[1][j] = wmma16(a1.v, vv[j].v, accT[1][j]);
        }
    }

    const int rbase = mBase + 32 * waveM + 8 * kh;
    const int cbase = nBase + 32 * waveN + r;
    for (int i = 0; i < 2; i++)
        for (int j = 0; j < 2; j++) {
            int col = cbase + 16 * j;
            for (int v = 0; v < 8; v++) {
                int row = rbase + 16 * i + v;
                float sv = accS[i][j][v];
                float g = (sv / (1.f + __expf(-sv))) * accT[i][j][v];
                out[(size_t)row * N + col] = (_Float16)g;
            }
        }
}

// ---------------------------------------------------------------- flash attention
// qkv: (T, 3C) f16. Per wave: one 16-row query tile of one head.
__global__ __launch_bounds__(256) void attn_wmma(const _Float16* __restrict__ qkv,
                                                 _Float16* __restrict__ y16) {
    __shared__ __align__(16) _Float16 pSh[8 * 16 * 32];  // per-wave P staging
    const int lane = threadIdx.x & 31;
    const int wave = threadIdx.x >> 5;
    const int head = blockIdx.y;
    const int qBase = blockIdx.x * 128 + wave * 16;
    const int r  = lane & 15;
    const int kh = lane >> 4;
    const size_t rs = (size_t)C3;

    const _Float16* Qp = qkv + (size_t)head * D_HEAD;
    const _Float16* Kp = qkv + C_DIM + (size_t)head * D_HEAD;
    const _Float16* Vp = qkv + 2 * C_DIM + (size_t)head * D_HEAD;

    Frag qa0, qa1;
    const _Float16* qrow = Qp + (size_t)(qBase + r) * rs;
    qa0.q[0] = *(const uint4*)(qrow + kh * 8);
    qa0.q[1] = *(const uint4*)(qrow + 16 + kh * 8);
    qa1.q[0] = *(const uint4*)(qrow + 32 + kh * 8);
    qa1.q[1] = *(const uint4*)(qrow + 48 + kh * 8);

    float Mx[8], Ls[8];
    v8f O[4];
    v8f zero = {};
    for (int v = 0; v < 8; v++) { Mx[v] = -3e38f; Ls[v] = 0.f; }
    for (int t = 0; t < 4; t++) O[t] = zero;

    _Float16* pW = pSh + wave * (16 * 32);
    const float scale = 0.125f;  // 1/sqrt(64)

    for (int s0 = 0; s0 < qBase + 16; s0 += 32) {
        v8f sf0 = zero, sf1 = zero;
        {
            Frag b0, b1;
            const _Float16* kr0 = Kp + (size_t)(s0 + r) * rs + kh * 8;
            const _Float16* kr1 = Kp + (size_t)(s0 + 16 + r) * rs + kh * 8;
            b0.q[0] = *(const uint4*)(kr0);
            b0.q[1] = *(const uint4*)(kr0 + 16);
            b1.q[0] = *(const uint4*)(kr1);
            b1.q[1] = *(const uint4*)(kr1 + 16);
            sf0 = wmma16(qa0.v, b0.v, sf0);
            sf1 = wmma16(qa0.v, b1.v, sf1);
            b0.q[0] = *(const uint4*)(kr0 + 32);
            b0.q[1] = *(const uint4*)(kr0 + 48);
            b1.q[0] = *(const uint4*)(kr1 + 32);
            b1.q[1] = *(const uint4*)(kr1 + 48);
            sf0 = wmma16(qa1.v, b0.v, sf0);
            sf1 = wmma16(qa1.v, b1.v, sf1);
        }

        float p0[8], p1[8];
        for (int v = 0; v < 8; v++) {
            int row = qBase + 8 * kh + v;
            int c0 = s0 + r, c1 = s0 + 16 + r;
            float a0 = (c0 <= row) ? sf0[v] * scale : -3e38f;
            float a1 = (c1 <= row) ? sf1[v] * scale : -3e38f;
            float mx = fmaxf(a0, a1);
            for (int m = 8; m >= 1; m >>= 1) mx = fmaxf(mx, __shfl_xor(mx, m, 32));
            float newM = fmaxf(Mx[v], mx);
            float alpha = __expf(Mx[v] - newM);
            float e0 = __expf(a0 - newM);
            float e1 = __expf(a1 - newM);
            float ls = e0 + e1;
            for (int m = 8; m >= 1; m >>= 1) ls += __shfl_xor(ls, m, 32);
            Ls[v] = Ls[v] * alpha + ls;
            Mx[v] = newM;
            for (int t = 0; t < 4; t++) O[t][v] *= alpha;
            p0[v] = e0; p1[v] = e1;
        }

        for (int v = 0; v < 8; v++) {
            pW[(8 * kh + v) * 32 + r]      = (_Float16)p0[v];
            pW[(8 * kh + v) * 32 + 16 + r] = (_Float16)p1[v];
        }
        Frag pa;
        pa.q[0] = *(const uint4*)&pW[r * 32 + kh * 8];
        pa.q[1] = *(const uint4*)&pW[r * 32 + 16 + kh * 8];

        for (int dt = 0; dt < 4; dt++) {
            Frag vb;
            const _Float16* vcol = Vp + dt * 16 + r;
            for (int i = 0; i < 16; i++)
                vb.h[i] = vcol[(size_t)(s0 + kh * 8 + i + (i >= 8 ? 8 : 0)) * rs];
            O[dt] = wmma16(pa.v, vb.v, O[dt]);
        }
    }

    for (int dt = 0; dt < 4; dt++)
        for (int v = 0; v < 8; v++) {
            int row = qBase + 8 * kh + v;
            y16[(size_t)row * C_DIM + head * D_HEAD + dt * 16 + r] =
                (_Float16)(O[dt][v] / Ls[v]);
        }
}

// ---------------------------------------------------------------- launch
extern "C" void kernel_launch(void* const* d_in, const int* in_sizes, int n_in,
                              void* d_out, int out_size, void* d_ws, size_t ws_size,
                              hipStream_t stream) {
    const float* x     = (const float*)d_in[0];
    const float* ln1_w = (const float*)d_in[1];
    const float* ln1_b = (const float*)d_in[2];
    const float* Wqkv  = (const float*)d_in[3];
    const float* bqkv  = (const float*)d_in[4];
    const float* Wo    = (const float*)d_in[5];
    const float* bo    = (const float*)d_in[6];
    const float* ln2_w = (const float*)d_in[7];
    const float* ln2_b = (const float*)d_in[8];
    const float* Wfc   = (const float*)d_in[9];
    const float* bfc   = (const float*)d_in[10];
    const float* Wsw   = (const float*)d_in[11];
    const float* Vsw   = (const float*)d_in[12];
    const float* Wproj = (const float*)d_in[13];
    const float* bproj = (const float*)d_in[14];
    float* out = (float*)d_out;

    char* ws = (char*)d_ws;
    size_t off = 0;
    auto alloc = [&](size_t bytes) -> char* {
        char* p = ws + off;
        off = (off + bytes + 255) & ~(size_t)255;
        return p;
    };

    _Float16* h16    = (_Float16*)alloc((size_t)T_SEQ * C_DIM * 2);
    _Float16* qkv16  = (_Float16*)alloc((size_t)T_SEQ * C3 * 2);
    _Float16* y16    = (_Float16*)alloc((size_t)T_SEQ * C_DIM * 2);
    float*    x1     = (float*)   alloc((size_t)T_SEQ * C_DIM * 4);
    _Float16* h2_16  = (_Float16*)alloc((size_t)T_SEQ * C_DIM * 2);
    _Float16* fbuf   = (_Float16*)alloc((size_t)T_SEQ * C4 * 2);
    _Float16* g16    = (_Float16*)alloc((size_t)T_SEQ * C4 * 2);
    _Float16* wqkv16 = (_Float16*)alloc((size_t)C3 * C_DIM * 2);
    _Float16* wo16   = (_Float16*)alloc((size_t)C_DIM * C_DIM * 2);
    _Float16* wfc16  = (_Float16*)alloc((size_t)C4 * C_DIM * 2);
    _Float16* wswT   = (_Float16*)alloc((size_t)C4 * C4 * 2);
    _Float16* vswT   = (_Float16*)alloc((size_t)C4 * C4 * 2);
    _Float16* wpr16  = (_Float16*)alloc((size_t)C_DIM * C4 * 2);

    auto cgrid = [](int n) { return (n + 255) / 256 > 8192 ? 8192 : (n + 255) / 256; };

    convert_f32_to_f16<<<cgrid(C3 * C_DIM), 256, 0, stream>>>(Wqkv, wqkv16, C3 * C_DIM);
    convert_f32_to_f16<<<cgrid(C_DIM * C_DIM), 256, 0, stream>>>(Wo, wo16, C_DIM * C_DIM);
    convert_f32_to_f16<<<cgrid(C4 * C_DIM), 256, 0, stream>>>(Wfc, wfc16, C4 * C_DIM);
    convert_f32_to_f16<<<cgrid(C_DIM * C4), 256, 0, stream>>>(Wproj, wpr16, C_DIM * C4);
    transpose_f32_to_f16<<<cgrid(C4 * C4), 256, 0, stream>>>(Wsw, wswT, C4, C4);
    transpose_f32_to_f16<<<cgrid(C4 * C4), 256, 0, stream>>>(Vsw, vswT, C4, C4);

    // 1) h = ln1(x)
    ln_f16<<<T_SEQ, 256, 0, stream>>>(x, ln1_w, ln1_b, h16);

    // 2) qkv = h @ Wqkv^T + bqkv
    gemm_wmma_f16<<<dim3(C3 / 128, T_SEQ / 128), 256, 0, stream>>>(
        h16, wqkv16, bqkv, nullptr, nullptr, qkv16, T_SEQ, C3, C_DIM);

    // 3) y = causal attention
    attn_wmma<<<dim3(T_SEQ / 128, H_HEADS), 256, 0, stream>>>(qkv16, y16);

    // 4) x1 = x + y @ Wo^T + bo
    gemm_wmma_f16<<<dim3(C_DIM / 128, T_SEQ / 128), 256, 0, stream>>>(
        y16, wo16, bo, x, x1, nullptr, T_SEQ, C_DIM, C_DIM);

    // 5) h2 = ln2(x1)
    ln_f16<<<T_SEQ, 256, 0, stream>>>(x1, ln2_w, ln2_b, h2_16);

    // 6) f = h2 @ Wfc^T + bfc
    gemm_wmma_f16<<<dim3(C4 / 128, T_SEQ / 128), 256, 0, stream>>>(
        h2_16, wfc16, bfc, nullptr, nullptr, fbuf, T_SEQ, C4, C_DIM);

    // 7) g = silu(f @ Wsw) * (f @ Vsw)
    gemm_swiglu_wmma<<<dim3(C4 / 64, T_SEQ / 128), 256, 0, stream>>>(
        fbuf, wswT, vswT, g16, T_SEQ, C4, C4);

    // 8) out = x1 + g @ Wproj^T + bproj
    gemm_wmma_f16<<<dim3(C_DIM / 128, T_SEQ / 128), 256, 0, stream>>>(
        g16, wpr16, bproj, x1, out, nullptr, T_SEQ, C_DIM, C4);
}
